// Model_24670292148349
// MI455X (gfx1250) — matrix operand order repaired and checked by reading, not compiled, not run on hardware
//
#include <hip/hip_runtime.h>
#include <stdint.h>

// ---------------------------------------------------------------------------
// MI455X (gfx1250) BiLSTM autoencoder.
//  - persistent-RNN scans: recurrent+input weights live in VGPRs (wave32,
//    up to 1024 VGPRs/wave), cell state lives in WMMA accumulators,
//    h-tile is exchanged through LDS each timestep.
//  - all GEMMs via v_wmma_f32_16x16x32_bf16 (f32 accumulate).
//  - per-step activation tile staged global->LDS via the Tensor Data Mover
//    (tensor_load_to_lds + s_wait_tensorcnt) when the builtin is available.
// ---------------------------------------------------------------------------

typedef __attribute__((ext_vector_type(16))) __bf16        v16bf;
typedef __attribute__((ext_vector_type(16))) unsigned short us16;
typedef __attribute__((ext_vector_type(8)))  float          v8f;
typedef __attribute__((ext_vector_type(4)))  unsigned int   v4ui;
typedef __attribute__((ext_vector_type(4)))  unsigned int   v4u;
typedef __attribute__((ext_vector_type(8)))  int            v8i;
typedef __attribute__((ext_vector_type(4)))  int            v4i;

#if defined(__has_builtin)
#  if __has_builtin(__builtin_amdgcn_tensor_load_to_lds)
#    define ASTRA_TDM 1
#  endif
#endif
#ifndef ASTRA_TDM
#  define ASTRA_TDM 0
#endif

#define DEVINL __device__ __forceinline__

DEVINL unsigned short f2bf(float f) {
    unsigned int u = __float_as_uint(f);
    u += 0x7FFFu + ((u >> 16) & 1u);
    return (unsigned short)(u >> 16);
}

union AB16 {           // all members trivially constructible (ext vectors)
    us16  us;
    v16bf bf;
    v4ui  q[2];
};

// ------------------------- fp32 -> bf16 pack kernel -------------------------
__global__ void k_cvt_bf16(const float* __restrict__ s,
                           unsigned short* __restrict__ d, int n) {
    for (int i = blockIdx.x * blockDim.x + threadIdx.x; i < n;
         i += gridDim.x * blockDim.x)
        d[i] = f2bf(s[i]);
}

// ------------------- zin = z @ W_d1 + b  (tiny one-shot GEMM) ---------------
__global__ void k_zin(const float* __restrict__ z,      // (B,128)
                      const float* __restrict__ W,      // (128,256)
                      const float* __restrict__ b,      // (256,)
                      float* __restrict__ out, int Bsz) // (B,256)
{
    int idx = blockIdx.x * blockDim.x + threadIdx.x;
    if (idx >= Bsz * 256) return;
    int row = idx >> 8, col = idx & 255;
    float s = b[col];
    const float* zr = z + row * 128;
#pragma unroll 8
    for (int k = 0; k < 128; ++k) s = fmaf(zr[k], W[k * 256 + col], s);
    out[idx] = s;
}

// ---------------------------------------------------------------------------
// Persistent LSTM scan.
//   grid = (B/16, 2)  -- blockIdx.y selects direction (0=fwd, 1=bwd).
//   U/16 waves; wave w owns hidden columns [16w,16w+16) and all 4 gates.
//   A-matrix (16 x K) staged in LDS: cols [0,IN)=x_t (TDM), [IN,IN+U)=h.
//   B-matrix ([W;U] as K x 4U bf16) resident in VGPRs for the whole scan.
// ---------------------------------------------------------------------------
template <int U, int IN, bool HAS_RC, bool WRITE_H, bool WRITE_FINAL>
__global__ __launch_bounds__(2 * U, 1)
void k_lstm_scan(const unsigned short* __restrict__ X,   // bf16 (B,T,IN) or null
                 const unsigned short* __restrict__ Wf,  // bf16 (K,4U) fwd
                 const unsigned short* __restrict__ Wb,  // bf16 (K,4U) bwd
                 const float* __restrict__ biasf, const float* __restrict__ biasb,
                 const float* __restrict__ rcf,  const float* __restrict__ rcb,
                 unsigned short* __restrict__ hout, int hstride, int hoff_f, int hoff_b,
                 float* __restrict__ fin, int foff_f, int foff_b,
                 int T)
{
    constexpr int K   = IN + U;      // fused reduction dim (mult of 32)
    constexpr int NK  = K / 32;      // wmma k-steps
    constexpr int TPB = 2 * U;       // threads per block

    __shared__ unsigned short Ast[16 * K];   // sole shared var -> LDS offset 0

    const int tid  = threadIdx.x;
    const int lane = tid & 31;
    const int wv   = tid >> 5;
    const int hi   = lane >> 4;      // half-wave select (WMMA layouts)
    const int ln   = lane & 15;
    const int n0   = wv * 16;        // this wave's hidden slice base
    const int dir  = blockIdx.y;
    const bool rev = (dir != 0);
    const int rowbase = blockIdx.x * 16;

    const unsigned short* W = dir ? Wb : Wf;
    const float* bias = dir ? biasb : biasf;
    const float* rc   = dir ? rcb : rcf;
    const int hoff    = dir ? hoff_b : hoff_f;
    const int foff    = dir ? foff_b : foff_f;

    // ---- preload B operands (weights) into registers, WMMA 32x16 layout ----
    // us16 element j of lane L holds Wcat[kt*32 + 16*(L>>4) + j][col]
    us16 Breg[4][NK];
#pragma unroll
    for (int g = 0; g < 4; ++g) {
        const int col = g * U + n0 + ln;
#pragma unroll
        for (int kt = 0; kt < NK; ++kt) {
            AB16 bb;
#pragma unroll
            for (int j = 0; j < 16; ++j)
                bb.us[j] = W[(size_t)(kt * 32 + hi * 16 + j) * (4 * U) + col];
            Breg[g][kt] = bb.us;
        }
    }

    // per-gate additive constant (bias, or per-row const input for decoder-1)
    float cadd[4][8];
#pragma unroll
    for (int g = 0; g < 4; ++g) {
        if constexpr (HAS_RC) {
#pragma unroll
            for (int v = 0; v < 8; ++v)
                cadd[g][v] = rc[(size_t)(rowbase + v + 8 * hi) * (4 * U) +
                                g * U + n0 + ln];
        } else {
            float bv = bias[g * U + n0 + ln];
#pragma unroll
            for (int v = 0; v < 8; ++v) cadd[g][v] = bv;
        }
    }

    // h(0) = 0 in LDS, c(0) = 0 in accum registers
    for (int i = tid; i < 16 * K; i += TPB) Ast[i] = 0;
    v8f cst;
#pragma unroll
    for (int v = 0; v < 8; ++v) cst[v] = 0.0f;
    __syncthreads();

    for (int tau = 0; tau < T; ++tau) {
        const int t = rev ? (T - 1 - tau) : tau;

        if constexpr (IN > 0) {
#if ASTRA_TDM
            // ---- Tensor Data Mover: DMA 16 x IN bf16 tile into Ast[:,0:IN]
            // LDS padding interleaves rows at stride K (pad U/2 dwords per row)
            if (tid < 32) {
                constexpr unsigned int pic = (IN == 64) ? 4u
                                           : (IN == 128) ? 5u : 6u; // log2(IN)-2
                constexpr unsigned int pam = (unsigned int)(U / 2 - 1);
                uint64_t ga = (uint64_t)(uintptr_t)(X + ((size_t)rowbase * T + t) * IN);
                v4u g0;
                g0[0] = 1u;                                   // count = 1 (valid)
                g0[1] = 0u;                                   // lds_addr = offset of Ast
                g0[2] = (unsigned int)(ga & 0xFFFFFFFFu);
                g0[3] = (unsigned int)((ga >> 32) & 0x1FFFFFFu) | (2u << 30); // type=2
                v8i g1;
                g1[0] = (int)((1u << 16) | (1u << 20) | (pic << 22) | (pam << 25));
                g1[1] = (int)((unsigned int)(IN & 0xFFFF) << 16);   // tensor_dim0 lo
                g1[2] = (int)(((unsigned int)IN >> 16) | (16u << 16)); // dim0 hi|dim1 lo
                g1[3] = (int)((unsigned int)IN << 16);              // dim1 hi | tile_dim0
                g1[4] = (int)16;                                    // tile_dim1=16
                unsigned long long s0 = (unsigned long long)T * (unsigned long long)IN;
                g1[5] = (int)(s0 & 0xFFFFFFFFu);                    // dim0 stride lo
                g1[6] = (int)((s0 >> 32) & 0xFFFFu);                // stride hi
                g1[7] = 0;
                v4i g2; g2[0] = 0; g2[1] = 0; g2[2] = 0; g2[3] = 0;
                v4i g3; g3[0] = 0; g3[1] = 0; g3[2] = 0; g3[3] = 0;
                v8i g4; g4[0] = 0; g4[1] = 0; g4[2] = 0; g4[3] = 0;
                g4[4] = 0; g4[5] = 0; g4[6] = 0; g4[7] = 0;
                // clang-23 / therock-10.0 6-arg form (extra int32x8 group)
                __builtin_amdgcn_tensor_load_to_lds(g0, g1, g2, g3, g4, 0);
                __builtin_amdgcn_s_wait_tensorcnt(0);
            }
#else
            // cooperative fallback: 16B chunks, rows strided by T*IN in global
            constexpr int CH = IN / 8;
            for (int i = tid; i < 16 * CH; i += TPB) {
                int r = i / CH, c = i % CH;
                const v4ui* src = (const v4ui*)(X + ((size_t)(rowbase + r) * T + t) * IN) + c;
                *(v4ui*)&Ast[r * K + c * 8] = *src;
            }
#endif
        }
        __syncthreads();   // x_t staged, h(t-1) visible

        // ---- z = [x_t | h] @ [W;U] + const : 4 gate tiles via WMMA ----
        v8f acc[4];
#pragma unroll
        for (int g = 0; g < 4; ++g)
#pragma unroll
            for (int v = 0; v < 8; ++v) acc[g][v] = cadd[g][v];

#pragma unroll
        for (int kt = 0; kt < NK; ++kt) {
            AB16 a;   // A 16x32 bf16 layout: two b128 LDS loads
            const unsigned short* ap = &Ast[(size_t)ln * K + kt * 32 + 16 * hi];
            a.q[0] = *(const v4ui*)ap;
            a.q[1] = *(const v4ui*)(ap + 8);
            v16bf av = a.bf;
#pragma unroll
            for (int g = 0; g < 4; ++g) {
                acc[g] = __builtin_amdgcn_wmma_f32_16x16x32_bf16(
                    false, av, false, __builtin_bit_cast(v16bf, Breg[g][kt]),
                    (short)0, acc[g], false, false);
            }
        }

        // ---- gate math (i,f,o=sigmoid, g=relu; h=o*relu(c)) ----
        float hv[8];
#pragma unroll
        for (int v = 0; v < 8; ++v) {
            float ig = 1.0f / (1.0f + __expf(-acc[0][v]));
            float fg = 1.0f / (1.0f + __expf(-acc[1][v]));
            float gg = fmaxf(acc[2][v], 0.0f);
            float og = 1.0f / (1.0f + __expf(-acc[3][v]));
            float cn = fg * cst[v] + ig * gg;
            cst[v]   = cn;
            hv[v]    = og * fmaxf(cn, 0.0f);
        }

        __syncthreads();   // all waves done reading Ast before h overwrite

#pragma unroll
        for (int v = 0; v < 8; ++v) {
            unsigned short hb = f2bf(hv[v]);
            int m = v + 8 * hi;
            Ast[(size_t)m * K + IN + n0 + ln] = hb;
            if constexpr (WRITE_H)
                hout[((size_t)(rowbase + m) * T + t) * hstride + hoff + n0 + ln] = hb;
        }
        if constexpr (WRITE_FINAL) {
            if (tau == T - 1) {
#pragma unroll
                for (int v = 0; v < 8; ++v)
                    fin[(size_t)(rowbase + v + 8 * hi) * (2 * U) + foff + n0 + ln] = hv[v];
            }
        }
    }
}

// ------------------- final dense: (B*T,256) @ (256,64) + bd -----------------
__global__ __launch_bounds__(128, 1)
void k_dense(const unsigned short* __restrict__ A,   // bf16 (M,256)
             const unsigned short* __restrict__ Wb,  // bf16 (256,64)
             const float* __restrict__ bd,           // (64,)
             float* __restrict__ out, int M)
{
    __shared__ unsigned short Ast[16 * 256];
    const int tid = threadIdx.x, lane = tid & 31, wv = tid >> 5;
    const int hi = lane >> 4, ln = lane & 15, n0 = wv * 16;

    us16 Breg[8];
#pragma unroll
    for (int kt = 0; kt < 8; ++kt) {
        AB16 bb;
#pragma unroll
        for (int j = 0; j < 16; ++j)
            bb.us[j] = Wb[(size_t)(kt * 32 + hi * 16 + j) * 64 + n0 + ln];
        Breg[kt] = bb.us;
    }
    const float bv = bd[n0 + ln];

    for (int mt = blockIdx.x; mt < M / 16; mt += gridDim.x) {
        const v4ui* src = (const v4ui*)(A + (size_t)mt * 16 * 256);
        for (int i = tid; i < (16 * 256 / 8); i += 128)
            *(v4ui*)&Ast[i * 8] = src[i];
        __syncthreads();

        v8f acc;
#pragma unroll
        for (int v = 0; v < 8; ++v) acc[v] = bv;
#pragma unroll
        for (int kt = 0; kt < 8; ++kt) {
            AB16 a;
            const unsigned short* ap = &Ast[(size_t)ln * 256 + kt * 32 + 16 * hi];
            a.q[0] = *(const v4ui*)ap;
            a.q[1] = *(const v4ui*)(ap + 8);
            acc = __builtin_amdgcn_wmma_f32_16x16x32_bf16(
                false, a.bf, false, __builtin_bit_cast(v16bf, Breg[kt]),
                (short)0, acc, false, false);
        }
#pragma unroll
        for (int v = 0; v < 8; ++v)
            out[(size_t)(mt * 16 + v + 8 * hi) * 64 + n0 + ln] = acc[v];
        __syncthreads();
    }
}

// ---------------------------------------------------------------------------
extern "C" void kernel_launch(void* const* d_in, const int* in_sizes, int n_in,
                              void* d_out, int out_size, void* d_ws, size_t ws_size,
                              hipStream_t stream)
{
    (void)in_sizes; (void)n_in; (void)out_size; (void)ws_size;
    const int B = 512, T = 256, F = 64;

    const float* x     = (const float*)d_in[0];
    const float* e1fW  = (const float*)d_in[1];
    const float* e1fU  = (const float*)d_in[2];
    const float* e1fb  = (const float*)d_in[3];
    const float* e1bW  = (const float*)d_in[4];
    const float* e1bU  = (const float*)d_in[5];
    const float* e1bb  = (const float*)d_in[6];
    const float* e2fW  = (const float*)d_in[7];
    const float* e2fU  = (const float*)d_in[8];
    const float* e2fb  = (const float*)d_in[9];
    const float* e2bW  = (const float*)d_in[10];
    const float* e2bU  = (const float*)d_in[11];
    const float* e2bb  = (const float*)d_in[12];
    const float* d1fW  = (const float*)d_in[13];
    const float* d1fU  = (const float*)d_in[14];
    const float* d1fb  = (const float*)d_in[15];
    const float* d1bW  = (const float*)d_in[16];
    const float* d1bU  = (const float*)d_in[17];
    const float* d1bb  = (const float*)d_in[18];
    const float* d2fW  = (const float*)d_in[19];
    const float* d2fU  = (const float*)d_in[20];
    const float* d2fb  = (const float*)d_in[21];
    const float* d2bW  = (const float*)d_in[22];
    const float* d2bU  = (const float*)d_in[23];
    const float* d2bb  = (const float*)d_in[24];
    const float* Wd    = (const float*)d_in[25];
    const float* bd    = (const float*)d_in[26];

    // ---- workspace carve (256B aligned) ----
    size_t off = 0;
    char* wsb = (char*)d_ws;
    auto carve = [&](size_t bytes) -> void* {
        void* p = wsb + off;
        off += (bytes + 255) & ~(size_t)255;
        return p;
    };
    unsigned short* xbf   = (unsigned short*)carve((size_t)B * T * F * 2);
    unsigned short* h1    = (unsigned short*)carve((size_t)B * T * 256 * 2);
    unsigned short* d1o   = (unsigned short*)carve((size_t)B * T * 128 * 2);
    unsigned short* d2o   = (unsigned short*)carve((size_t)B * T * 256 * 2);
    float* zbuf  = (float*)carve((size_t)B * 128 * 4);
    float* zin_f = (float*)carve((size_t)B * 256 * 4);
    float* zin_b = (float*)carve((size_t)B * 256 * 4);
    unsigned short* Wc_e1f = (unsigned short*)carve((size_t)192 * 512 * 2);
    unsigned short* Wc_e1b = (unsigned short*)carve((size_t)192 * 512 * 2);
    unsigned short* Wc_e2f = (unsigned short*)carve((size_t)320 * 256 * 2);
    unsigned short* Wc_e2b = (unsigned short*)carve((size_t)320 * 256 * 2);
    unsigned short* Uc_d1f = (unsigned short*)carve((size_t)64 * 256 * 2);
    unsigned short* Uc_d1b = (unsigned short*)carve((size_t)64 * 256 * 2);
    unsigned short* Wc_d2f = (unsigned short*)carve((size_t)256 * 512 * 2);
    unsigned short* Wc_d2b = (unsigned short*)carve((size_t)256 * 512 * 2);
    unsigned short* Wdbf   = (unsigned short*)carve((size_t)256 * 64 * 2);

    auto cvt = [&](const float* src, unsigned short* dst, int n) {
        int blocks = (n + 255) / 256;
        if (blocks > 2048) blocks = 2048;
        k_cvt_bf16<<<blocks, 256, 0, stream>>>(src, dst, n);
    };

    // ---- pack inputs/weights to bf16 (Wcat = [W;U] row-major K x 4u) ----
    cvt(x, xbf, B * T * F);
    cvt(e1fW, Wc_e1f, 64 * 512);  cvt(e1fU, Wc_e1f + 64 * 512, 128 * 512);
    cvt(e1bW, Wc_e1b, 64 * 512);  cvt(e1bU, Wc_e1b + 64 * 512, 128 * 512);
    cvt(e2fW, Wc_e2f, 256 * 256); cvt(e2fU, Wc_e2f + 256 * 256, 64 * 256);
    cvt(e2bW, Wc_e2b, 256 * 256); cvt(e2bU, Wc_e2b + 256 * 256, 64 * 256);
    cvt(d1fU, Uc_d1f, 64 * 256);
    cvt(d1bU, Uc_d1b, 64 * 256);
    cvt(d2fW, Wc_d2f, 128 * 512); cvt(d2fU, Wc_d2f + 128 * 512, 128 * 512);
    cvt(d2bW, Wc_d2b, 128 * 512); cvt(d2bU, Wc_d2b + 128 * 512, 128 * 512);
    cvt(Wd, Wdbf, 256 * 64);

    dim3 grid(B / 16, 2);

    // encoder-1: BiLSTM(u=128, in=64) -> h1 (B,T,256)
    k_lstm_scan<128, 64, false, true, false><<<grid, 256, 0, stream>>>(
        xbf, Wc_e1f, Wc_e1b, e1fb, e1bb, nullptr, nullptr,
        h1, 256, 0, 128, nullptr, 0, 0, T);

    // encoder-2: two LSTMs(u=64, in=256), keep final states -> z (B,128)
    k_lstm_scan<64, 256, false, false, true><<<grid, 128, 0, stream>>>(
        h1, Wc_e2f, Wc_e2b, e2fb, e2bb, nullptr, nullptr,
        nullptr, 0, 0, 0, zbuf, 0, 64, T);

    // decoder-1 input projection (time-invariant): zin = z @ W_d1 + b
    k_zin<<<(B * 256 + 255) / 256, 256, 0, stream>>>(zbuf, d1fW, d1fb, zin_f, B);
    k_zin<<<(B * 256 + 255) / 256, 256, 0, stream>>>(zbuf, d1bW, d1bb, zin_b, B);

    // decoder-1: BiLSTM(u=64, constant input) -> d1o (B,T,128)
    k_lstm_scan<64, 0, true, true, false><<<grid, 128, 0, stream>>>(
        nullptr, Uc_d1f, Uc_d1b, nullptr, nullptr, zin_f, zin_b,
        d1o, 128, 0, 64, nullptr, 0, 0, T);

    // decoder-2: BiLSTM(u=128, in=128) -> d2o (B,T,256)
    k_lstm_scan<128, 128, false, true, false><<<grid, 256, 0, stream>>>(
        d1o, Wc_d2f, Wc_d2b, d2fb, d2bb, nullptr, nullptr,
        d2o, 256, 0, 128, nullptr, 0, 0, T);

    // final dense: (B*T,256) @ (256,64) + bd -> out (B,T,64) fp32
    k_dense<<<512, 128, 0, stream>>>(d2o, Wdbf, bd, (float*)d_out, B * T);
}
